// Attention_60206851555616
// MI455X (gfx1250) — matrix operand order repaired and checked by reading, not compiled, not run on hardware
//
#include <hip/hip_runtime.h>

// ---------------------------------------------------------------------------
// Types matching CDNA5 WMMA / TDM builtin signatures
// ---------------------------------------------------------------------------
typedef __attribute__((ext_vector_type(16))) __bf16 v16bf;
typedef __attribute__((ext_vector_type(8)))  __bf16 v8bf;
typedef __attribute__((ext_vector_type(8)))  float  v8f;
typedef __attribute__((ext_vector_type(4)))  float  v4f;
typedef __attribute__((ext_vector_type(4)))  unsigned int u32x4;
typedef __attribute__((ext_vector_type(8)))  int    i32x8;
typedef __attribute__((ext_vector_type(4)))  int    i32x4;

#define DEV __device__ __forceinline__

DEV v16bf cat16(v8bf lo, v8bf hi) {
  return __builtin_shufflevector(lo, hi, 0,1,2,3,4,5,6,7,8,9,10,11,12,13,14,15);
}
DEV v8bf ld8(const __bf16* p) { return *(const v8bf*)p; }

DEV v8f wmma_bf16(v16bf a, v16bf b, v8f c) {
  return __builtin_amdgcn_wmma_f32_16x16x32_bf16(false, a, false, b, (short)0, c,
                                                 false, false);
}

// ---------------------------------------------------------------------------
// Problem constants
// ---------------------------------------------------------------------------
constexpr int B  = 8;
constexpr int SA = 1024;
constexpr int SV = 2048;
constexpr int D  = 512;
constexpr int DQ = 128;

// workspace layout (bytes), everything 256B-aligned by construction
constexpr size_t OFF_AB  = 0;                                     // a  -> bf16 [B*SA][512]
constexpr size_t OFF_VB  = OFF_AB  + (size_t)B*SA*D*2;            // v  -> bf16 [B*SV][512]
constexpr size_t OFF_WQT = OFF_VB  + (size_t)B*SV*D*2;            // WqT bf16 [128][512]
constexpr size_t OFF_WKT = OFF_WQT + (size_t)DQ*D*2;              // WkT bf16 [128][512]
constexpr size_t OFF_WVT = OFF_WKT + (size_t)DQ*D*2;              // WvT bf16 [512][512]
constexpr size_t OFF_QB  = OFF_WVT + (size_t)D*D*2;               // q   bf16 [B*SA][128]
constexpr size_t OFF_KB  = OFF_QB  + (size_t)B*SA*DQ*2;           // k   bf16 [B*SV][128]
constexpr size_t OFF_VT  = OFF_KB  + (size_t)B*SV*DQ*2;           // valT bf16 [B][512][SV]

// LDS layout of flash_attn (dynamic)
constexpr unsigned LDS_PROBS_BYTES = 16u * SV * 2u;        // 64 KB
constexpr unsigned LDS_KBUF_BYTES  = 2u * 16u * DQ * 2u;   // 8 KB double buffer
constexpr unsigned LDS_TOTAL_BYTES = LDS_PROBS_BYTES + LDS_KBUF_BYTES;

// ---------------------------------------------------------------------------
// f32 -> bf16 elementwise convert (8 elems / thread, b128-friendly)
// ---------------------------------------------------------------------------
__global__ void cvt_bf16(const float* __restrict__ src, __bf16* __restrict__ dst,
                         int n8) {
  int i = blockIdx.x * blockDim.x + threadIdx.x;
  if (i >= n8) return;
  const v4f* s = (const v4f*)src + (size_t)i * 2;
  v4f x0 = s[0], x1 = s[1];
  v8bf o;
  o[0]=(__bf16)x0[0]; o[1]=(__bf16)x0[1]; o[2]=(__bf16)x0[2]; o[3]=(__bf16)x0[3];
  o[4]=(__bf16)x1[0]; o[5]=(__bf16)x1[1]; o[6]=(__bf16)x1[2]; o[7]=(__bf16)x1[3];
  *(v8bf*)(dst + (size_t)i * 8) = o;
}

// W [512][N] f32  ->  WT [N][512] bf16
__global__ void transp_bf16(const float* __restrict__ W, __bf16* __restrict__ WT,
                            int N) {
  int i = blockIdx.x * blockDim.x + threadIdx.x;
  if (i >= N * 512) return;
  int n = i >> 9, k = i & 511;
  WT[(size_t)n * 512 + k] = (__bf16)W[(size_t)k * N + n];
}

// ---------------------------------------------------------------------------
// C[M][N] (+bias) = A[M][512] x WT[N][512]^T ; bf16 in, bf16 out.
// One wave = one 16x64 tile: A-fragment loaded once per K-step and reused
// across 4 B-fragments / 4 accumulators (2x FLOP/byte vs 16x16 tiles).
// transposed==1 stores val as [B][512][SV] (output transpose, 2048 rows/batch)
// ---------------------------------------------------------------------------
__global__ void gemm_bias_bf16(const __bf16* __restrict__ A,
                               const __bf16* __restrict__ WT,
                               const float*  __restrict__ bias,
                               __bf16* __restrict__ out,
                               int M, int N, int transposed) {
  int gwave = (int)((blockIdx.x * blockDim.x + threadIdx.x) >> 5);
  int lane  = threadIdx.x & 31;
  int tilesN = N >> 6;                       // 64-wide N tiles
  int tiles  = (M >> 4) * tilesN;
  if (gwave >= tiles) return;                // wave-uniform: EXEC stays all-1s
  int tm = gwave / tilesN, tn = gwave - tm * tilesN;
  int l16 = lane & 15, half = lane >> 4;

  const __bf16* Ar = A + (size_t)(tm * 16 + l16) * 512;   // A-frag row

  int ncol[4];
  const __bf16* Wr[4];
  v8f acc[4];
#pragma unroll
  for (int j = 0; j < 4; ++j) {
    ncol[j] = tn * 64 + j * 16 + l16;
    Wr[j]   = WT + (size_t)ncol[j] * 512;
    float bval = bias[ncol[j]];
#pragma unroll
    for (int r = 0; r < 8; ++r) acc[j][r] = bval;
  }

#pragma unroll 2
  for (int ks = 0; ks < 16; ++ks) {
    int kb = ks * 32;
    v16bf af = cat16(ld8(Ar + kb + half * 16),
                     ld8(Ar + kb + half * 16 + 8));
#pragma unroll
    for (int j = 0; j < 4; ++j) {
      v16bf bf_ = cat16(ld8(Wr[j] + kb + half * 16),
                        ld8(Wr[j] + kb + half * 16 + 8));
      acc[j] = wmma_bf16(af, bf_, acc[j]);
    }
  }

  if (!transposed) {
#pragma unroll
    for (int j = 0; j < 4; ++j)
#pragma unroll
      for (int r = 0; r < 8; ++r) {
        int orow = tm * 16 + half * 8 + r;
        out[(size_t)orow * N + ncol[j]] = (__bf16)acc[j][r];
      }
  } else {
#pragma unroll
    for (int j = 0; j < 4; ++j)
#pragma unroll
      for (int r = 0; r < 8; ++r) {
        int orow = tm * 16 + half * 8 + r;       // global v row in [0, B*SV)
        int bb = orow >> 11, mloc = orow & 2047; // SV == 2048
        out[((size_t)bb * 512 + ncol[j]) * SV + mloc] = (__bf16)acc[j][r];
      }
  }
}

// ---------------------------------------------------------------------------
// TDM: stage one 16x128 bf16 k-tile into LDS via tensor_load_to_lds.
// Descriptor bit layout per CDNA5 ISA ch.8 (group0: count/lds/global/type,
// group1: data_size/dims/tile/stride). Wave-uniform operands.
// ---------------------------------------------------------------------------
DEV void tdm_load_ktile(const __bf16* gsrc, const __bf16* ldst) {
  unsigned long long ga = (unsigned long long)(const void*)gsrc;
  u32x4 g0;
  g0[0] = 1u;                                   // count=1, is_restore=0, gather=0
  g0[1] = (unsigned)(unsigned long long)(const void*)ldst; // LDS byte address
  g0[2] = (unsigned)(ga & 0xffffffffu);         // global_addr[31:0]
  g0[3] = (unsigned)((ga >> 32) & 0x01ffffffu)  // global_addr[56:32]
        | (2u << 30);                           // type = 2 ("image")
  i32x8 g1;
  g1[0] = 1 << 16;          // data_size = 1 (2 bytes/elem)
  g1[1] = DQ << 16;         // tensor_dim0[15:0] = 128
  g1[2] = SV << 16;         // tensor_dim0 hi = 0 | tensor_dim1[15:0] = 2048
  g1[3] = DQ << 16;         // tensor_dim1 hi = 0 | tile_dim0 = 128
  g1[4] = 16;               // tile_dim1 = 16, tile_dim2 = 0
  g1[5] = DQ;               // tensor_dim0_stride = 128 elems
  g1[6] = 0;
  g1[7] = 0;
  i32x4 z4 = {0, 0, 0, 0};
#if __clang_major__ >= 23
  i32x8 z8 = {0, 0, 0, 0, 0, 0, 0, 0};
  __builtin_amdgcn_tensor_load_to_lds(g0, g1, z4, z4, z8, 0);
#else
  __builtin_amdgcn_tensor_load_to_lds(g0, g1, z4, z4, 0);
#endif
}

// ---------------------------------------------------------------------------
// Fused attention: one wave (blockDim=32) per 16-query tile.
//  pass1: S = q k^T (WMMA, k staged into LDS by the Tensor Data Mover with
//         double buffering + s_wait_tensorcnt), per-lane online max/sum,
//         shfl_xor row reduction
//  pass2: probs = exp(S - m)/l -> 64 KB LDS (16 x 2048 bf16), k re-read from
//         global (L2-hot by now)
//  pass3: out = probs @ valT (WMMA, A-frags via ds_load_b128)
// ---------------------------------------------------------------------------
__global__ void flash_attn(const __bf16* __restrict__ qb,
                           const __bf16* __restrict__ kb,
                           const __bf16* __restrict__ vtb,
                           float* __restrict__ out) {
  extern __shared__ char smem[];
  __bf16* probs = (__bf16*)smem;                          // [16][2048]
  __bf16* kst   = (__bf16*)(smem + LDS_PROBS_BYTES);      // [2][16][128]

  int wid = blockIdx.x;               // 0 .. B*(SA/16)-1
  int b   = wid >> 6;                 // 64 query tiles / batch
  int qt  = wid & 63;
  int lane = threadIdx.x & 31;
  int l16 = lane & 15, half = lane >> 4;

  // q A-fragments for all 4 K-steps (d = 0..127), reused by both passes
  const __bf16* qr = qb + ((size_t)b * SA + qt * 16 + l16) * DQ;
  v16bf qf[4];
#pragma unroll
  for (int ks = 0; ks < 4; ++ks)
    qf[ks] = cat16(ld8(qr + ks * 32 + half * 16),
                   ld8(qr + ks * 32 + half * 16 + 8));

  const __bf16* kbase = kb + (size_t)b * SV * DQ;

  // ---- pass 1: TDM-staged k tiles + per-lane online softmax stats ----------
  float m[8], l[8];
#pragma unroll
  for (int r = 0; r < 8; ++r) { m[r] = -3.0e38f; l[r] = 0.f; }

  tdm_load_ktile(kbase, kst);                       // prologue: tile 0
  for (int kt = 0; kt < SV / 16; ++kt) {
    if (kt + 1 < SV / 16) {                         // wave-uniform branch
      tdm_load_ktile(kbase + (size_t)(kt + 1) * 16 * DQ,
                     kst + ((kt + 1) & 1) * 16 * DQ);
      __builtin_amdgcn_s_wait_tensorcnt(1);         // tile kt complete
    } else {
      __builtin_amdgcn_s_wait_tensorcnt(0);
    }
    const __bf16* kr = kst + (kt & 1) * 16 * DQ + (size_t)l16 * DQ;
    v8f s = {};
#pragma unroll
    for (int ks = 0; ks < 4; ++ks) {
      v16bf kf = cat16(*(const v8bf*)(kr + ks * 32 + half * 16),
                       *(const v8bf*)(kr + ks * 32 + half * 16 + 8));
      s = wmma_bf16(qf[ks], kf, s);
    }
#pragma unroll
    for (int r = 0; r < 8; ++r) {
      float mn = fmaxf(m[r], s[r]);
      l[r] = l[r] * __expf(m[r] - mn) + __expf(s[r] - mn);
      m[r] = mn;
    }
  }
  // reduce across the 16 lanes of each half (rows r+8*half live there)
#pragma unroll
  for (int off = 1; off < 16; off <<= 1) {
#pragma unroll
    for (int r = 0; r < 8; ++r) {
      float mo = __shfl_xor(m[r], off, 32);
      float lo = __shfl_xor(l[r], off, 32);
      float mn = fmaxf(m[r], mo);
      l[r] = l[r] * __expf(m[r] - mn) + lo * __expf(mo - mn);
      m[r] = mn;
    }
  }
  float inv[8];
#pragma unroll
  for (int r = 0; r < 8; ++r) inv[r] = 1.0f / l[r];

  // ---- pass 2: normalized probabilities into LDS ---------------------------
  for (int kt = 0; kt < SV / 16; ++kt) {
    const __bf16* kr = kbase + (size_t)(kt * 16 + l16) * DQ;
    v8f s = {};
#pragma unroll
    for (int ks = 0; ks < 4; ++ks) {
      v16bf kf = cat16(ld8(kr + ks * 32 + half * 16),
                       ld8(kr + ks * 32 + half * 16 + 8));
      s = wmma_bf16(qf[ks], kf, s);
    }
#pragma unroll
    for (int r = 0; r < 8; ++r) {
      float p = __expf(s[r] - m[r]) * inv[r];
      probs[(size_t)(half * 8 + r) * SV + kt * 16 + l16] = (__bf16)p;
    }
  }
  __syncthreads();

  // ---- pass 3: out = probs @ valT -----------------------------------------
  const __bf16* vt = vtb + (size_t)b * D * SV;
  float* orow = out + ((size_t)b * SA + qt * 16) * D;

  for (int ng = 0; ng < 8; ++ng) {           // 8 groups x 4 n-chunks = 512 cols
    v8f acc[4] = {};
    for (int kt2 = 0; kt2 < SV / 32; ++kt2) {
      int kk = kt2 * 32;
      v16bf pf = cat16(*(const v8bf*)(probs + (size_t)l16 * SV + kk + half * 16),
                       *(const v8bf*)(probs + (size_t)l16 * SV + kk + half * 16 + 8));
#pragma unroll
      for (int j = 0; j < 4; ++j) {
        const __bf16* vr = vt + (size_t)(ng * 64 + j * 16 + l16) * SV + kk + half * 16;
        v16bf vf = cat16(ld8(vr), ld8(vr + 8));
        acc[j] = wmma_bf16(pf, vf, acc[j]);
      }
    }
#pragma unroll
    for (int j = 0; j < 4; ++j) {
#pragma unroll
      for (int r = 0; r < 8; ++r) {
        int row = half * 8 + r;
        orow[(size_t)row * D + ng * 64 + j * 16 + l16] = acc[j][r];
      }
    }
  }
}

// ---------------------------------------------------------------------------
// Host-side launch
// ---------------------------------------------------------------------------
extern "C" void kernel_launch(void* const* d_in, const int* in_sizes, int n_in,
                              void* d_out, int out_size, void* d_ws, size_t ws_size,
                              hipStream_t stream) {
  const float* a  = (const float*)d_in[0];
  const float* v  = (const float*)d_in[1];
  const float* Wq = (const float*)d_in[2];
  const float* bq = (const float*)d_in[3];
  const float* Wk = (const float*)d_in[4];
  const float* bk = (const float*)d_in[5];
  const float* Wv = (const float*)d_in[6];
  const float* bv = (const float*)d_in[7];

  char* ws = (char*)d_ws;
  __bf16* ab  = (__bf16*)(ws + OFF_AB);
  __bf16* vb  = (__bf16*)(ws + OFF_VB);
  __bf16* WqT = (__bf16*)(ws + OFF_WQT);
  __bf16* WkT = (__bf16*)(ws + OFF_WKT);
  __bf16* WvT = (__bf16*)(ws + OFF_WVT);
  __bf16* qbb = (__bf16*)(ws + OFF_QB);
  __bf16* kbb = (__bf16*)(ws + OFF_KB);
  __bf16* vtb = (__bf16*)(ws + OFF_VT);

  // 1) convert activations + weights to bf16 (weights transposed to [N][512])
  {
    int n8 = B * SA * D / 8;
    cvt_bf16<<<(n8 + 255) / 256, 256, 0, stream>>>(a, ab, n8);
  }
  {
    int n8 = B * SV * D / 8;
    cvt_bf16<<<(n8 + 255) / 256, 256, 0, stream>>>(v, vb, n8);
  }
  transp_bf16<<<(DQ * 512 + 255) / 256, 256, 0, stream>>>(Wq, WqT, DQ);
  transp_bf16<<<(DQ * 512 + 255) / 256, 256, 0, stream>>>(Wk, WkT, DQ);
  transp_bf16<<<(D  * 512 + 255) / 256, 256, 0, stream>>>(Wv, WvT, D);

  // 2) projections (WMMA GEMMs, 16x64 tiles/wave); val written as [B][512][SV]
  {
    int tiles = (B * SA / 16) * (DQ / 64);        // 1024 waves
    gemm_bias_bf16<<<(tiles * 32 + 255) / 256, 256, 0, stream>>>(
        ab, WqT, bq, qbb, B * SA, DQ, 0);
  }
  {
    int tiles = (B * SV / 16) * (DQ / 64);        // 2048 waves
    gemm_bias_bf16<<<(tiles * 32 + 255) / 256, 256, 0, stream>>>(
        vb, WkT, bk, kbb, B * SV, DQ, 0);
  }
  {
    int tiles = (B * SV / 16) * (D / 64);         // 8192 waves
    gemm_bias_bf16<<<(tiles * 32 + 255) / 256, 256, 0, stream>>>(
        vb, WvT, bv, vtb, B * SV, D, 1);
  }

  // 3) fused flash attention: 512 single-wave workgroups, 72 KB LDS each
  flash_attn<<<B * (SA / 16), 32, LDS_TOTAL_BYTES, stream>>>(
      qbb, kbb, vtb, (float*)d_out);
}
